// Stem_12008728560264
// MI455X (gfx1250) — compile-verified, matrix operand order checked by reading, not run on hardware
//
#include <hip/hip_runtime.h>
#include <hip/hip_bf16.h>

typedef __attribute__((ext_vector_type(16))) __bf16 v16bf;
typedef __attribute__((ext_vector_type(8)))  __bf16 v8bf;
typedef __attribute__((ext_vector_type(8)))  float  v8f;

// ---------------------------------------------------------------- helpers
__device__ __forceinline__ float sigm_f(float x) {
    return 1.0f / (1.0f + __expf(-x));
}
__device__ __forceinline__ float tanh_f(float x) {
    x = fminf(fmaxf(x, -15.0f), 15.0f);
    float e = __expf(2.0f * x);
    return (e - 1.0f) / (e + 1.0f);
}
// load a 16x32 (bf16) WMMA operand fragment for this lane.
// row_ptr: base of the 16-element "row" this lane owns; kbase already
// includes half*8 (ISA 16-bit operand layout: two 8-elem runs, +0 and +16).
__device__ __forceinline__ v16bf load_frag(const __bf16* __restrict__ row_ptr, int kbase) {
    v8bf lo = *(const v8bf*)(row_ptr + kbase);
    v8bf hi = *(const v8bf*)(row_ptr + kbase + 16);
    v16bf a;
#pragma unroll
    for (int i = 0; i < 8; ++i) { a[i] = lo[i]; a[8 + i] = hi[i]; }
    return a;
}
__device__ __forceinline__ v16bf load_frag_lds(const __bf16* row_ptr, int kbase) {
    v8bf lo = *(const v8bf*)(row_ptr + kbase);
    v8bf hi = *(const v8bf*)(row_ptr + kbase + 16);
    v16bf a;
#pragma unroll
    for (int i = 0; i < 8; ++i) { a[i] = lo[i]; a[8 + i] = hi[i]; }
    return a;
}

// ---------------------------------------------------------------- fp32 -> bf16
__global__ void f2bf_kernel(const float* __restrict__ src, __bf16* __restrict__ dst, int n) {
    int i = blockIdx.x * blockDim.x + threadIdx.x;
    if (i < n) dst[i] = (__bf16)src[i];
}
__global__ void ex_copy_kernel(const unsigned char* __restrict__ s,
                               unsigned char* __restrict__ d, int n) {
    int i = blockIdx.x * blockDim.x + threadIdx.x;
    if (i < n) d[i] = s[i] ? 1 : 0;
}

// ---------------------------------------------------------------- G = X @ W^T + b
// X: (R,512) bf16, W: (512,512) bf16 (rows are output cols), G: (2,R,512) f32.
// Block = 256 thr (8 waves), one 16-row tile, all 512 cols (wave -> 4 n-tiles).
__global__ __launch_bounds__(256)
void gemm_ih_kernel(const __bf16* __restrict__ X,
                    const __bf16* __restrict__ Wf, const __bf16* __restrict__ Wb,
                    const float* __restrict__ bf, const float* __restrict__ bb,
                    float* __restrict__ G, int R) {
    const int dir = blockIdx.y;
    const __bf16* W = dir ? Wb : Wf;
    const float* bias = dir ? bb : bf;
    float* Gd = G + (size_t)dir * R * 512;

    const int wave = threadIdx.x >> 5;
    const int lane = threadIdx.x & 31;
    const int half = lane >> 4;
    const int n    = lane & 15;

    const int row0   = blockIdx.x * 16;
    const int ntile0 = wave * 4;          // 4 tiles of 16 cols each

    v8f acc[4] = {};
    const __bf16* Arow = X + (size_t)(row0 + n) * 512;

    for (int kt = 0; kt < 16; ++kt) {
        const int kbase = kt * 32 + half * 8;
        v16bf a = load_frag(Arow, kbase);
        __builtin_prefetch(Arow + kbase + 64, 0, 1);
#pragma unroll
        for (int e = 0; e < 4; ++e) {
            const int j = (ntile0 + e) * 16 + n;      // col of output = row of W
            v16bf b = load_frag(W + (size_t)j * 512, kbase);
            acc[e] = __builtin_amdgcn_wmma_f32_16x16x32_bf16(
                false, a, false, b, (short)0, acc[e], false, false);
        }
    }
#pragma unroll
    for (int e = 0; e < 4; ++e) {
        const int col = (ntile0 + e) * 16 + n;
        const float bv = bias[col];
#pragma unroll
        for (int r = 0; r < 8; ++r) {
            const int m = half * 8 + r;
            Gd[(size_t)(row0 + m) * 512 + col] = acc[e][r] + bv;
        }
    }
}

// ---------------------------------------------------------------- recurrence
// grid = 2 (dir), block = 1024 (32 waves). Sequential over timesteps.
// LDS: Whh 512x128 bf16 (128K) | hg 32x512 f32 (64K) | c 32x128 f32 (16K)
//      | h 32x128 bf16 (8K)  => 216K of the WGP's 320K pool.
#define LSTM_SMEM_BYTES (512 * 128 * 2 + 32 * 512 * 4 + 32 * 128 * 4 + 32 * 128 * 2)
__global__ __launch_bounds__(1024)
void lstm_seq_kernel(const float* __restrict__ G,
                     const __bf16* __restrict__ WhhF, const __bf16* __restrict__ WhhB,
                     const float* __restrict__ h0, const float* __restrict__ c0,
                     float* __restrict__ hseq, int len) {
    extern __shared__ char smem[];
    __bf16* whh_l = (__bf16*)smem;                                        // 512*128
    float*  hg    = (float*)(smem + 512 * 128 * 2);                       // 32*512
    float*  cst   = (float*)(smem + 512 * 128 * 2 + 32 * 512 * 4);        // 32*128
    __bf16* hbf   = (__bf16*)(smem + 512 * 128 * 2 + 32 * 512 * 4 + 32 * 128 * 4);

    const int dir = blockIdx.x;
    const int R = 32 * len;
    const float* Gd = G + (size_t)dir * R * 512;
    float* Hd = hseq + (size_t)dir * R * 128;
    const __bf16* Whh = dir ? WhhB : WhhF;

    const int tid = threadIdx.x;
    // preload Whh into LDS (128KB) as uint4 copies
    {
        const uint4* src = (const uint4*)Whh;
        uint4* dst = (uint4*)whh_l;
#pragma unroll 2
        for (int i = tid; i < 512 * 128 * 2 / 16; i += 1024) dst[i] = src[i];
    }
    // broadcast initial state: h = tanh(h0[dir]), c = c0[dir]
#pragma unroll
    for (int k = 0; k < 4; ++k) {
        const int idx = tid + k * 1024;
        const int j = idx & 127;
        hbf[idx] = (__bf16)tanh_f(h0[dir * 128 + j]);
        cst[idx] = c0[dir * 128 + j];
    }
    __syncthreads();

    const int wave = tid >> 5, lane = tid & 31;
    const int half = lane >> 4, n = lane & 15;
    const int mtile  = wave >> 4;         // 0..1
    const int ntile0 = (wave & 15) * 2;   // 2 n-tiles per wave
    const __bf16* Ar = hbf + (mtile * 16 + n) * 128;

    for (int step = 0; step < len; ++step) {
        const int t = dir ? (len - 1 - step) : step;

        // hg = h @ Whh^T  (32x512, K=128), operands fully in LDS
        v8f acc[2] = {};
#pragma unroll
        for (int kt = 0; kt < 4; ++kt) {
            const int kbase = kt * 32 + half * 8;
            v16bf a = load_frag_lds(Ar, kbase);
#pragma unroll
            for (int e = 0; e < 2; ++e) {
                const int j = (ntile0 + e) * 16 + n;
                v16bf b = load_frag_lds(whh_l + j * 128, kbase);
                acc[e] = __builtin_amdgcn_wmma_f32_16x16x32_bf16(
                    false, a, false, b, (short)0, acc[e], false, false);
            }
        }
#pragma unroll
        for (int e = 0; e < 2; ++e) {
            const int col = (ntile0 + e) * 16 + n;
#pragma unroll
            for (int r = 0; r < 8; ++r)
                hg[(mtile * 16 + half * 8 + r) * 512 + col] = acc[e][r];
        }
        __syncthreads();

        // fused gate nonlinearity + state update (4 (b,j) pairs per thread)
#pragma unroll
        for (int k = 0; k < 4; ++k) {
            const int idx = tid + k * 1024;
            const int b = idx >> 7, j = idx & 127;
            const float* gr = Gd + (size_t)(b * len + t) * 512;
            const float gi = gr[j      ] + hg[b * 512 + j      ];
            const float gf = gr[j + 128] + hg[b * 512 + j + 128];
            const float gg = gr[j + 256] + hg[b * 512 + j + 256];
            const float go = gr[j + 384] + hg[b * 512 + j + 384];
            const float cv = sigm_f(gf) * cst[idx] + sigm_f(gi) * tanh_f(gg);
            cst[idx] = cv;
            const float hv = sigm_f(go) * tanh_f(cv);
            Hd[(size_t)(b * len + t) * 128 + j] = hv;
            hbf[idx] = (__bf16)hv;
        }
        __syncthreads();
    }
}

// ---------------------------------------------------------------- orient = [hf|hb] @ Wo^T + bo
__global__ void orient_kernel(const float* __restrict__ hseq, const float* __restrict__ Wo,
                              const float* __restrict__ bo, float* __restrict__ tmp_or,
                              float* __restrict__ out_or, int len, int seg) {
    const int p = blockIdx.x * blockDim.x + threadIdx.x;  // b*len + t
    const int R = 32 * len;
    if (p >= R) return;
    const float4* hf4 = (const float4*)(hseq + (size_t)p * 128);
    const float4* hb4 = (const float4*)(hseq + (size_t)(R + p) * 128);
    const float4* Wo4 = (const float4*)Wo;
    float s = bo[0];
#pragma unroll 4
    for (int j = 0; j < 32; ++j) {
        const float4 h = hf4[j], w = Wo4[j];
        s += h.x * w.x + h.y * w.y + h.z * w.z + h.w * w.w;
    }
#pragma unroll 4
    for (int j = 0; j < 32; ++j) {
        const float4 h = hb4[j], w = Wo4[32 + j];
        s += h.x * w.x + h.y * w.y + h.z * w.z + h.w * w.w;
    }
    tmp_or[p] = s;
    const int b = p / len, t = p - b * len;
    out_or[(size_t)b * 2080 + seg + t] = s;
}

// ---------------------------------------------------------------- output + merge ('add' combinator)
__global__ void merge_kernel(const float* __restrict__ uh, const float* __restrict__ tmp_or,
                             const unsigned char* __restrict__ ex,
                             float* __restrict__ uh_next, unsigned char* __restrict__ ex_next,
                             float* __restrict__ out_unit, float* __restrict__ out_exist,
                             int len, int seg) {
    const int idx = blockIdx.x * blockDim.x + threadIdx.x;
    const int total = 32 * len * 128;  // (b, t, d4) with D/4 = 128
    if (idx >= total) return;
    const int d4 = idx & 127;
    const int t  = (idx >> 7) % len;
    const int b  = idx / (128 * len);

    const float4* u4 = (const float4*)uh;
    const float4 u = u4[(size_t)(b * len + t) * 128 + d4];
    ((float4*)out_unit)[((size_t)b * 2080 + seg + t) * 128 + d4] = u;

    const bool e_t = ex[b * len + t] != 0;
    if (d4 == 0)
        out_exist[(size_t)b * 2080 + seg + t] = e_t ? 1.0f : 0.0f;

    if (t < len - 1) {
        const bool r_t  = tmp_or[b * len + t] > 0.0f;
        const bool r_t1 = tmp_or[b * len + t + 1] > 0.0f;
        const bool e_t1 = ex[b * len + t + 1] != 0;
        const bool lc = r_t && e_t;
        const bool rc = (!r_t1) && e_t1;
        const float4 u1 = u4[(size_t)(b * len + t + 1) * 128 + d4];
        float4 r;
        r.x = (lc ? u.x : 0.0f) + (rc ? u1.x : 0.0f);
        r.y = (lc ? u.y : 0.0f) + (rc ? u1.y : 0.0f);
        r.z = (lc ? u.z : 0.0f) + (rc ? u1.z : 0.0f);
        r.w = (lc ? u.w : 0.0f) + (rc ? u1.w : 0.0f);
        ((float4*)uh_next)[(size_t)(b * (len - 1) + t) * 128 + d4] = r;
        if (d4 == 0) ex_next[b * (len - 1) + t] = (lc || rc) ? 1 : 0;
    }
}

// ---------------------------------------------------------------- host
extern "C" void kernel_launch(void* const* d_in, const int* in_sizes, int n_in,
                              void* d_out, int out_size, void* d_ws, size_t ws_size,
                              hipStream_t stream) {
    (void)in_sizes; (void)n_in; (void)out_size; (void)ws_size;
    const unsigned char* ex_in = (const unsigned char*)d_in[0];
    const float* uh_in = (const float*)d_in[1];
    const float* Wih_f = (const float*)d_in[2];
    const float* Whh_f = (const float*)d_in[3];
    const float* b_f   = (const float*)d_in[4];
    const float* Wih_b = (const float*)d_in[5];
    const float* Whh_b = (const float*)d_in[6];
    const float* b_b   = (const float*)d_in[7];
    const float* Wo    = (const float*)d_in[8];
    const float* bo    = (const float*)d_in[9];
    const float* h0    = (const float*)d_in[10];
    const float* c0    = (const float*)d_in[11];

    const int B = 32, L = 64, D = 512, T = 2080;

    // allow 216KB dynamic LDS for the recurrent kernel (WGP pool is 320KB)
    (void)hipFuncSetAttribute(reinterpret_cast<const void*>(&lstm_seq_kernel),
                              hipFuncAttributeMaxDynamicSharedMemorySize,
                              LSTM_SMEM_BYTES);

    // workspace carve (256B aligned)
    char* w = (char*)d_ws;
    auto carve = [&](size_t bytes) {
        char* p = w; w += (bytes + 255) & ~(size_t)255; return p;
    };
    __bf16* wih_bf = (__bf16*)carve((size_t)2 * 512 * 512 * 2);
    __bf16* whh_bf = (__bf16*)carve((size_t)2 * 512 * 128 * 2);
    __bf16* x_bf   = (__bf16*)carve((size_t)B * L * D * 2);
    float*  G      = (float*)carve((size_t)2 * B * L * 512 * 4);
    float*  hseq   = (float*)carve((size_t)2 * B * L * 128 * 4);
    float*  uhA    = (float*)carve((size_t)B * L * D * 4);
    float*  uhB    = (float*)carve((size_t)B * L * D * 4);
    float*  tmpo   = (float*)carve((size_t)B * L * 4);
    unsigned char* exA = (unsigned char*)carve(B * L);
    unsigned char* exB = (unsigned char*)carve(B * L);

    float* out_unit = (float*)d_out;                       // (B,T,512)
    float* out_or   = out_unit + (size_t)B * T * D;        // (B,T)
    float* out_ex   = out_or + (size_t)B * T;              // (B,T)

    // per-call weight conversion to bf16 (deterministic, cheap)
    f2bf_kernel<<<(512 * 512 + 255) / 256, 256, 0, stream>>>(Wih_f, wih_bf, 512 * 512);
    f2bf_kernel<<<(512 * 512 + 255) / 256, 256, 0, stream>>>(Wih_b, wih_bf + 512 * 512, 512 * 512);
    f2bf_kernel<<<(512 * 128 + 255) / 256, 256, 0, stream>>>(Whh_f, whh_bf, 512 * 128);
    f2bf_kernel<<<(512 * 128 + 255) / 256, 256, 0, stream>>>(Whh_b, whh_bf + 512 * 128, 512 * 128);
    ex_copy_kernel<<<(B * L + 255) / 256, 256, 0, stream>>>(ex_in, exA, B * L);

    const float* cur_uh = uh_in;
    const unsigned char* cur_ex = exA;

    for (int len = L; len >= 1; --len) {
        const int R = B * len;
        const int seg = len * (len - 1) / 2;

        f2bf_kernel<<<(R * 512 + 255) / 256, 256, 0, stream>>>(cur_uh, x_bf, R * 512);
        gemm_ih_kernel<<<dim3(R / 16, 2), 256, 0, stream>>>(
            x_bf, wih_bf, wih_bf + 512 * 512, b_f, b_b, G, R);
        lstm_seq_kernel<<<2, 1024, LSTM_SMEM_BYTES, stream>>>(
            G, whh_bf, whh_bf + 512 * 128, h0, c0, hseq, len);
        orient_kernel<<<(R + 255) / 256, 256, 0, stream>>>(
            hseq, Wo, bo, tmpo, out_or, len, seg);

        float* next_uh = (cur_uh == uhA) ? uhB : uhA;
        unsigned char* next_ex = (cur_ex == exA) ? exB : exA;
        merge_kernel<<<(B * len * 128 + 255) / 256, 256, 0, stream>>>(
            cur_uh, tmpo, cur_ex, next_uh, next_ex, out_unit, out_ex, len, seg);
        cur_uh = next_uh;
        cur_ex = next_ex;
    }
}
